// SS2D_60902636257675
// MI455X (gfx1250) — matrix operand order repaired and checked by reading, not compiled, not run on hardware
//
#include <hip/hip_runtime.h>

// ---------------------------------------------------------------------------
// SS2D (bidirectional dense-state Mamba scan) for gfx1250 / MI455X.
//   B=8, H=W=128, E=192, K_CONV=4.  All GEMM work on v_wmma_f32_16x16x32_bf16.
//
// Pipeline (one stream, order = dependency):
//   xcvt : x fp32 -> bf16 (read by both directions' convs)
//   prep : A = -exp(A_log) -> bf16 row-major (== B-operand transpose);
//          conv_w -> bf16 [k][n][e] (B-operand transpose)
//   conv(dir) : u = conv1d(x) via 4x6 WMMA per wave; x window async-staged in LDS
//   scan(dir) : h_t = A h_{t-1} + u_t via 6 WMMA per wave; A weight-stationary
//               in 48 VGPRs; h ping-pongs in LDS; u register-pipelined.
// ---------------------------------------------------------------------------

#define Bc 8
#define Hc 128
#define Wc 128
#define Ec 192
#define Lc 128          // scan length (both directions: 128)
#define NSEQ 1024       // B*H == B*W
#define NT 12           // E/16 n-tiles
#define KT 6            // E/32 k-tiles

typedef __attribute__((ext_vector_type(16))) __bf16 v16bf;
typedef __attribute__((ext_vector_type(8)))  float  v8f;

union BF16Tile { uint4 q[2]; v16bf v; };

__device__ __forceinline__ unsigned short f2bf(float f) {
    union { float f; unsigned u; } v; v.f = f;
    unsigned r = v.u + 0x7FFFu + ((v.u >> 16) & 1u);   // round-to-nearest-even
    return (unsigned short)(r >> 16);
}
__device__ __forceinline__ float bf2f(unsigned short h) {
    union { unsigned u; float f; } v; v.u = ((unsigned)h) << 16; return v.f;
}

#if defined(__gfx1250__) && __has_builtin(__builtin_amdgcn_global_load_async_to_lds_b128) \
                         && __has_builtin(__builtin_amdgcn_s_wait_asynccnt)
#define HAVE_ASYNC_LDS 1
// builtin signature (from hipcc diagnostic): (v4i addrspace(1)*, v4i addrspace(3)*, imm, imm)
typedef int v4i_vs __attribute__((vector_size(16)));
typedef __attribute__((address_space(1))) v4i_vs* async_gptr;
typedef __attribute__((address_space(3))) v4i_vs* async_lptr;
#else
#define HAVE_ASYNC_LDS 0
#endif

// ---------------------------------------------------------------------------
// xcvt: x fp32 -> bf16, 8 elements per thread (coalesced b128 in / b128 out)
// ---------------------------------------------------------------------------
__global__ void ss2d_xcvt(const float* __restrict__ x, unsigned short* __restrict__ xbf,
                          int n8) {
    int i = blockIdx.x * blockDim.x + threadIdx.x;
    if (i >= n8) return;
    const float4* src = (const float4*)(x) + 2 * i;
    float4 a = src[0], b = src[1];
    union { unsigned short h[8]; uint4 q; } o;
    o.h[0] = f2bf(a.x); o.h[1] = f2bf(a.y); o.h[2] = f2bf(a.z); o.h[3] = f2bf(a.w);
    o.h[4] = f2bf(b.x); o.h[5] = f2bf(b.y); o.h[6] = f2bf(b.z); o.h[7] = f2bf(b.w);
    ((uint4*)xbf)[i] = o.q;
}

// ---------------------------------------------------------------------------
// prep: A_bf[dir][n][k] = bf16(-exp(A_log[n][k]))
//       CWt[dir][k][n][e] = bf16(conv_w[k][e][n])
// ---------------------------------------------------------------------------
__global__ void ss2d_prep(const float* __restrict__ hA, const float* __restrict__ wA,
                          const float* __restrict__ hCW, const float* __restrict__ wCW,
                          unsigned short* __restrict__ Abf,
                          unsigned short* __restrict__ CWt) {
    int i = blockIdx.x * blockDim.x + threadIdx.x;
    if (i < 2 * Ec * Ec) {
        const float* src = (i < Ec * Ec) ? hA : wA;
        int j = (i < Ec * Ec) ? i : i - Ec * Ec;
        Abf[i] = f2bf(-__expf(src[j]));
    }
    if (i < 2 * 4 * Ec * Ec) {
        const float* src = (i < 4 * Ec * Ec) ? hCW : wCW;
        int j = (i < 4 * Ec * Ec) ? i : i - 4 * Ec * Ec;
        int k = j / (Ec * Ec);
        int rem = j - k * Ec * Ec;
        int n = rem / Ec;
        int e = rem - n * Ec;
        CWt[i] = f2bf(src[(k * Ec + e) * Ec + n]);   // [k][cin=e][cout=n] -> [k][n][e]
    }
}

// ---------------------------------------------------------------------------
// conv: u[s][t][n] = bias[n] + sum_{k=0..3} x[s][t+k-1][:] @ W_k[:][n]
// One WG = one 16-position tile of one sequence; 12 waves = 12 N-tiles.
// 19x192 bf16 x-window staged in LDS (async-to-LDS when available).
// ---------------------------------------------------------------------------
__global__ __launch_bounds__(384) void ss2d_conv(const unsigned short* __restrict__ xbf,
                                                 const unsigned short* __restrict__ CWt,
                                                 const float* __restrict__ cbias,
                                                 unsigned short* __restrict__ U,
                                                 int dir) {
    __shared__ unsigned short xs[19 * Ec];

    const int tid = threadIdx.x;
    const int s   = blockIdx.x >> 3;          // sequence 0..1023
    const int t0  = (blockIdx.x & 7) * 16;    // position tile base

    int xbase, xstride;
    if (dir == 0) { xbase = s * (Wc * Ec);                       xstride = Ec; }
    else          { int b = s >> 7, w = s & 127;
                    xbase = b * (Hc * Wc * Ec) + w * Ec;         xstride = Wc * Ec; }

    // zero-fill boundary rows (t<0 / t>=L), only on first/last tile of sequence
    if (t0 == 0) {
        for (int e = tid; e < Ec; e += 384) xs[e] = 0;                       // row 0
    } else if (t0 == Lc - 16) {
        for (int e = tid; e < 2 * Ec; e += 384) xs[17 * Ec + e] = 0;         // rows 17,18
    }

    // stage valid rows: 456 chunks of 8 bf16 (16 B) each
    for (int c = tid; c < 19 * (Ec / 8); c += 384) {
        int row = c / (Ec / 8);
        int ce  = (c - row * (Ec / 8)) * 8;
        int t   = t0 + row - 1;
        if (t >= 0 && t < Lc) {
#if HAVE_ASYNC_LDS
            __builtin_amdgcn_global_load_async_to_lds_b128(
                (async_gptr)(xbf + xbase + t * xstride + ce),
                (async_lptr)(xs + row * Ec + ce),
                0, 0);
#else
            *(uint4*)(xs + row * Ec + ce) = *(const uint4*)(xbf + xbase + t * xstride + ce);
#endif
        }
    }
#if HAVE_ASYNC_LDS
    __builtin_amdgcn_s_wait_asynccnt(0);
#endif
    __syncthreads();

    const int wave = tid >> 5, lane = tid & 31;
    const int nb = wave * 16, nl = lane & 15, kh = lane >> 4;

    float bias = cbias[nb + nl];
    v8f acc;
    #pragma unroll
    for (int r = 0; r < 8; ++r) acc[r] = bias;

    #pragma unroll
    for (int k = 0; k < 4; ++k) {
        const unsigned short* wrow = CWt + (k * Ec + nb + nl) * Ec;
        const unsigned short* arow = xs + (nl + k) * Ec;   // t = t0 + m + k - 1
        #pragma unroll
        for (int kt = 0; kt < KT; ++kt) {
            BF16Tile a, b;
            int ce = kt * 32 + (kh << 3);      // A-op: K halves at +0/+16
            a.q[0] = *(const uint4*)(arow + ce);
            a.q[1] = *(const uint4*)(arow + ce + 16);
            int be = kt * 32 + (kh << 3);      // B-op: K halves at +0/+16
            b.q[0] = *(const uint4*)(wrow + be);
            b.q[1] = *(const uint4*)(wrow + be + 16);
            acc = __builtin_amdgcn_wmma_f32_16x16x32_bf16(false, a.v, false, b.v,
                                                          (short)0, acc, false, false);
        }
    }

    unsigned short* urow = U + (s * Lc) * Ec;
    #pragma unroll
    for (int r = 0; r < 8; ++r) {
        int m = r + (kh << 3);                 // C-tile: m = vgpr + 8*(lane>=16)
        urow[(t0 + m) * Ec + nb + nl] = f2bf(acc[r]);
    }
}

// ---------------------------------------------------------------------------
// scan: h_t = A h_{t-1} + u_t ; out_t (+)= h_t * D
// One WG = 16 sequences, 12 waves (one N-tile each). A-matrix B-operands are
// weight-stationary in 48 VGPRs/wave; h ping-pongs through LDS in bf16;
// u_{t+1} register-pipelined so global latency hides behind the WMMA chain.
// ---------------------------------------------------------------------------
__global__ __launch_bounds__(384) void ss2d_scan(const unsigned short* __restrict__ U,
                                                 const unsigned short* __restrict__ Abf,
                                                 const float* __restrict__ Dp,
                                                 float* __restrict__ out,
                                                 int dir, int addout) {
    __shared__ unsigned short hbuf[2][16 * Ec];

    const int tid  = threadIdx.x;
    const int wave = tid >> 5, lane = tid & 31;
    const int nb = wave * 16, nl = lane & 15, kh = lane >> 4;
    const int s0 = blockIdx.x * 16;

    for (int i = tid; i < 16 * Ec; i += 384) hbuf[0][i] = 0;   // h_{-1} = 0

    // weight-stationary B operands: rows of A (already the needed transpose)
    BF16Tile bA[KT];
    const unsigned short* arow = Abf + (nb + nl) * Ec;
    #pragma unroll
    for (int kt = 0; kt < KT; ++kt) {
        int be = kt * 32 + (kh << 3);
        bA[kt].q[0] = *(const uint4*)(arow + be);
        bA[kt].q[1] = *(const uint4*)(arow + be + 16);
    }
    const float dscale = Dp[nb + nl];

    int ubr[8], obr[8], ostride = (dir == 0) ? Ec : Wc * Ec;
    #pragma unroll
    for (int r = 0; r < 8; ++r) {
        int m = r + (kh << 3);
        int s = s0 + m;
        ubr[r] = s * (Lc * Ec) + nb + nl;
        if (dir == 0) obr[r] = s * (Wc * Ec) + nb + nl;
        else { int b = s >> 7, w = s & 127; obr[r] = b * (Hc * Wc * Ec) + w * Ec + nb + nl; }
    }

    // prime the u pipeline (t = 0)
    unsigned short ucur[8];
    #pragma unroll
    for (int r = 0; r < 8; ++r) ucur[r] = U[ubr[r]];

    __syncthreads();

    for (int t = 0; t < Lc; ++t) {
        const int p = t & 1, q = p ^ 1;

        // prefetch u_{t+1} into registers; latency hidden by WMMA chain below
        unsigned short unxt[8];
        if (t + 1 < Lc) {
            #pragma unroll
            for (int r = 0; r < 8; ++r) unxt[r] = U[ubr[r] + (t + 1) * Ec];
        }
        if (t + 8 < Lc) __builtin_prefetch(&U[ubr[0] + (t + 8) * Ec], 0, 0);

        v8f c;
        #pragma unroll
        for (int r = 0; r < 8; ++r) c[r] = bf2f(ucur[r]);      // C = u_t

        const unsigned short* hrow = &hbuf[p][nl * Ec];
        #pragma unroll
        for (int kt = 0; kt < KT; ++kt) {
            BF16Tile a;
            int ce = kt * 32 + (kh << 3);
            a.q[0] = *(const uint4*)(hrow + ce);
            a.q[1] = *(const uint4*)(hrow + ce + 16);
            c = __builtin_amdgcn_wmma_f32_16x16x32_bf16(false, a.v, false, bA[kt].v,
                                                        (short)0, c, false, false);
        }

        #pragma unroll
        for (int r = 0; r < 8; ++r) {
            int m = r + (kh << 3);
            hbuf[q][m * Ec + nb + nl] = f2bf(c[r]);            // h_t -> next step
            float ov = c[r] * dscale;
            int oi = obr[r] + t * ostride;
            if (addout) out[oi] += ov; else out[oi] = ov;
        }
        if (t + 1 < Lc) {
            #pragma unroll
            for (int r = 0; r < 8; ++r) ucur[r] = unxt[r];
        }
        __syncthreads();
    }
}

// ---------------------------------------------------------------------------
// launcher
// ---------------------------------------------------------------------------
extern "C" void kernel_launch(void* const* d_in, const int* in_sizes, int n_in,
                              void* d_out, int out_size, void* d_ws, size_t ws_size,
                              hipStream_t stream) {
    const float* x   = (const float*)d_in[0];
    const float* hCW = (const float*)d_in[3];
    const float* hCB = (const float*)d_in[4];
    const float* hA  = (const float*)d_in[5];
    const float* hD  = (const float*)d_in[6];
    const float* wCW = (const float*)d_in[9];
    const float* wCB = (const float*)d_in[10];
    const float* wA  = (const float*)d_in[11];
    const float* wD  = (const float*)d_in[12];
    float* out = (float*)d_out;

    unsigned char* ws = (unsigned char*)d_ws;
    size_t off = 0;
    unsigned short* Abf = (unsigned short*)(ws + off); off += (size_t)2 * Ec * Ec * 2;       // 144 KB
    unsigned short* CWt = (unsigned short*)(ws + off); off += (size_t)2 * 4 * Ec * Ec * 2;   // 576 KB
    unsigned short* xbf = (unsigned short*)(ws + off); off += (size_t)Bc * Hc * Wc * Ec * 2; // 48 MB
    unsigned short* U   = (unsigned short*)(ws + off);                                       // 48 MB (reused per dir)

    const int nelem = Bc * Hc * Wc * Ec;
    ss2d_xcvt<<<(nelem / 8 + 255) / 256, 256, 0, stream>>>(x, xbf, nelem / 8);
    ss2d_prep<<<(2 * 4 * Ec * Ec + 255) / 256, 256, 0, stream>>>(hA, wA, hCW, wCW, Abf, CWt);

    // horizontal: conv -> scan (out =)
    ss2d_conv<<<NSEQ * (Lc / 16), 384, 0, stream>>>(xbf, CWt, hCB, U, 0);
    ss2d_scan<<<NSEQ / 16, 384, 0, stream>>>(U, Abf, hD, out, 0, 0);

    // vertical: conv -> scan (out +=)   (stream order makes U reuse safe)
    ss2d_conv<<<NSEQ * (Lc / 16), 384, 0, stream>>>(xbf, CWt + 4 * Ec * Ec, wCB, U, 1);
    ss2d_scan<<<NSEQ / 16, 384, 0, stream>>>(U, Abf + Ec * Ec, wD, out, 1, 1);
}
